// VAE_79525614452823
// MI455X (gfx1250) — matrix operand order repaired and checked by reading, not compiled, not run on hardware
//
#include <hip/hip_runtime.h>
#include <hip/hip_bf16.h>

typedef __attribute__((ext_vector_type(16))) _Float16 v16h;
typedef __attribute__((ext_vector_type(8)))  _Float16 v8h;
typedef __attribute__((ext_vector_type(8)))  float    v8f;

#define WMMA_F16(A, Bm, C) \
  __builtin_amdgcn_wmma_f32_16x16x32_f16(false, (A), false, (Bm), (short)0, (C), false, false)

#define NB  16384   // batch
#define NT  100     // seq len
#define ND  5       // in/out dim
#define NH  32      // hidden
#define RECON_ELEMS (NB * NT * ND)  // 8,192,000

#define LOG2E  1.442695040888963f
#define LOG2E2 2.885390081777927f

// sigmoid with pre-scaled bias: sigm(x+b) = rcp(1 + 2^(fma(x,-log2e, nlb))), nlb = -log2e*b
__device__ __forceinline__ float sigm_fma(float x, float nlb) {
  return __builtin_amdgcn_rcpf(1.f + __builtin_amdgcn_exp2f(fmaf(x, -LOG2E, nlb)));
}

#if __has_builtin(__builtin_amdgcn_tanhf)
__device__ __forceinline__ float fast_tanh(float x) { return __builtin_amdgcn_tanhf(x); }
#else
__device__ __forceinline__ float fast_tanh(float x) {
  return 1.f - 2.f * __builtin_amdgcn_rcpf(__builtin_amdgcn_exp2f(LOG2E2 * x) + 1.f);
}
#endif

__device__ __forceinline__ float fast_exp(float x) {
  return __builtin_amdgcn_exp2f(LOG2E * x);
}
__device__ __forceinline__ float lrelu(float x) { return x > 0.f ? x : 0.01f * x; }

// B-matrix tile (32x16 f16) for gate tile `tile` from row-major W[4H x ldw]:
// B[k][n] = (k < kvalid) ? W[(tile*16+n)*ldw + k] : 0
__device__ __forceinline__ v16h load_B_gate(const float* __restrict__ W, int tile,
                                            int ldw, int kvalid, int lane) {
  int n  = lane & 15;
  int kb = (lane >> 4) * 8;
  v16h b;
#pragma unroll
  for (int e = 0; e < 16; ++e) {
    int k = kb + ((e < 8) ? e : e + 8);
    float v = (k < kvalid) ? W[(tile * 16 + n) * ldw + k] : 0.f;
    b[e] = (_Float16)v;
  }
  return b;
}

// Output-projection B (32x16): B[k][n] = (n<5) ? W_out[n*32+k] : 0
__device__ __forceinline__ v16h load_B_out(const float* __restrict__ Wout, int lane) {
  int n  = lane & 15;
  int kb = (lane >> 4) * 8;
  v16h b;
#pragma unroll
  for (int e = 0; e < 16; ++e)
    b[e] = (n < 5) ? (_Float16)Wout[n * NH + kb + ((e < 8) ? e : e + 8)] : (_Float16)0.f;
  return b;
}

// A-matrix (16x32 f16) from row-major f16 LDS tile [16][32].
__device__ __forceinline__ v16h loadA(const _Float16* __restrict__ lds, int lane) {
  int m  = lane & 15;
  int kb = (lane >> 4) * 8;
  v8h lo = *(const v8h*)(lds + m * NH + kb);
  v8h hi = *(const v8h*)(lds + m * NH + 16 + kb);
  v16h a;
#pragma unroll
  for (int e = 0; e < 8; ++e) { a[e] = lo[e]; a[8 + e] = hi[e]; }
  return a;
}

// A-matrix from row-major f32 LDS [16][ld] starting at column coloff.
__device__ __forceinline__ v16h loadA_f32(const float* __restrict__ lds, int lane,
                                          int ld, int coloff) {
  int m  = lane & 15;
  int kb = (lane >> 4) * 8;
  v16h a;
#pragma unroll
  for (int e = 0; e < 8; ++e) {
    a[e]     = (_Float16)lds[m * ld + coloff + kb + e];
    a[8 + e] = (_Float16)lds[m * ld + coloff + 16 + kb + e];
  }
  return a;
}

// One column-half LSTM cell update from 4 gate accumulators (i,f,g,o).
__device__ __forceinline__ void lstm_cell4(const v8f* acc, float nlb_i, float nlb_f,
                                           float b_g, float nlb_o, v8f& hh, v8f& cc) {
#pragma unroll
  for (int e = 0; e < 8; ++e) {
    float ig = sigm_fma(acc[0][e], nlb_i);
    float fg = sigm_fma(acc[1][e], nlb_f);
    float gg = fast_tanh(acc[2][e] + b_g);
    float og = sigm_fma(acc[3][e], nlb_o);
    float cn = fg * cc[e] + ig * gg;
    cc[e] = cn;
    hh[e] = og * fast_tanh(cn);
  }
}

__global__ __launch_bounds__(32) void vae_lstm_kernel(
    const float* __restrict__ x, const float* __restrict__ eps,
    const float* __restrict__ enc_Wih, const float* __restrict__ enc_Whh,
    const float* __restrict__ enc_bih, const float* __restrict__ enc_bhh,
    const float* __restrict__ W_mu, const float* __restrict__ b_mu,
    const float* __restrict__ W_lv, const float* __restrict__ b_lv,
    const float* __restrict__ W_l2h, const float* __restrict__ b_l2h,
    const float* __restrict__ W_l2h2, const float* __restrict__ b_l2h2,
    const float* __restrict__ start_token, const float* __restrict__ W_emb,
    const float* __restrict__ b_emb,
    const float* __restrict__ dec_Wih, const float* __restrict__ dec_Whh,
    const float* __restrict__ dec_bih, const float* __restrict__ dec_bhh,
    const float* __restrict__ W_out, const float* __restrict__ b_out,
    const float* __restrict__ W_seq, const float* __restrict__ b_seq,
    const float* __restrict__ W_seq2, const float* __restrict__ b_seq2,
    float* __restrict__ out) {
  // B-layout weight tiles live in LDS (keeps VGPR pressure <256, no MSB churn).
  __shared__ __align__(16) _Float16 wB[2][8][32][16];    // 16 KB: [ih/hh][tile][lane][e]
  __shared__ __align__(16) _Float16 wO[32][16];          // 1 KB output projection
  __shared__ __align__(16) _Float16 lds_h[2][16 * NH];   // 2 KB h staging (f16)
  __shared__ __align__(16) float    lds_f[16 * 64];      // 4 KB f32 scratch [h|c]

  const int lane    = threadIdx.x;        // 0..31
  const int n16     = lane & 15;
  const int rowh    = (lane >> 4) * 8;    // M offset for C/D layout
  const int rowbase = blockIdx.x * 16;

  v8f zero8;
#pragma unroll
  for (int e = 0; e < 8; ++e) zero8[e] = 0.f;

  // ================= Encoder =================
  float bias_e[8], nlb_e[8];
#pragma unroll
  for (int tile = 0; tile < 8; ++tile) {
    *(v16h*)&wB[0][tile][lane][0] = load_B_gate(enc_Wih, tile, ND, ND, lane);
    *(v16h*)&wB[1][tile][lane][0] = load_B_gate(enc_Whh, tile, NH, NH, lane);
    bias_e[tile] = enc_bih[tile * 16 + n16] + enc_bhh[tile * 16 + n16];
    nlb_e[tile]  = -LOG2E * bias_e[tile];
  }
  __syncthreads();

  v8f h[2], c[2];
#pragma unroll
  for (int p = 0; p < 2; ++p)
#pragma unroll
    for (int e = 0; e < 8; ++e) { h[p][e] = 0.f; c[p][e] = 0.f; }

  v16h a_h;
#pragma unroll
  for (int e = 0; e < 16; ++e) a_h[e] = (_Float16)0.f;

  for (int t = 0; t < NT; ++t) {
    // x_t as A-matrix: lanes 0-15 carry K=0..4, everything else zero (branchless)
    v16h a_x;
#pragma unroll
    for (int e = 0; e < 16; ++e) a_x[e] = (_Float16)0.f;
    {
      const float* xr = x + ((size_t)(rowbase + n16) * NT + t) * ND;
#pragma unroll
      for (int d = 0; d < ND; ++d) {
        float v = xr[d];
        a_x[d] = (_Float16)(lane < 16 ? v : 0.f);   // v_cndmask, no branch
      }
    }

    // Column-half group A (hidden cols 0-15): gate tiles 0,2,4,6 = i,f,g,o
    // Column-half group B (hidden cols 16-31): gate tiles 1,3,5,7
    v8f accA[4], accB[4];
#pragma unroll
    for (int q = 0; q < 4; ++q) {
      const int tile = 2 * q;
      v16h bw0 = *(const v16h*)&wB[0][tile][lane][0];
      v16h bw1 = *(const v16h*)&wB[1][tile][lane][0];
      accA[q] = WMMA_F16(a_x, bw0, zero8);
      accA[q] = WMMA_F16(a_h, bw1, accA[q]);
    }
#pragma unroll
    for (int q = 0; q < 4; ++q) {
      const int tile = 2 * q + 1;
      v16h bw0 = *(const v16h*)&wB[0][tile][lane][0];
      v16h bw1 = *(const v16h*)&wB[1][tile][lane][0];
      accB[q] = WMMA_F16(a_x, bw0, zero8);
      accB[q] = WMMA_F16(a_h, bw1, accB[q]);
    }
    // group-A cell math can co-execute with group-B WMMAs
    lstm_cell4(accA, nlb_e[0], nlb_e[2], bias_e[4], nlb_e[6], h[0], c[0]);
    lstm_cell4(accB, nlb_e[1], nlb_e[3], bias_e[5], nlb_e[7], h[1], c[1]);

    // C/D layout -> row-major f16 LDS -> A layout for next step
    _Float16* buf = lds_h[t & 1];
#pragma unroll
    for (int p = 0; p < 2; ++p)
#pragma unroll
      for (int v = 0; v < 8; ++v)
        buf[(v + rowh) * NH + p * 16 + n16] = (_Float16)h[p][v];
    __syncthreads();
    a_h = loadA(buf, lane);
  }

  // dump h|c as f32 rows for the scalar heads
#pragma unroll
  for (int p = 0; p < 2; ++p)
#pragma unroll
    for (int v = 0; v < 8; ++v) {
      lds_f[(v + rowh) * 64 + p * 16 + n16]      = h[p][v];
      lds_f[(v + rowh) * 64 + 32 + p * 16 + n16] = c[p][v];
    }
  __syncthreads();

  // ---- per-row heads: mu, log_var, z, num, decoder init (lanes 0-15, 1 row each)
  if (lane < 16) {
    const float* hc = lds_f + lane * 64;
    float mu = b_mu[0], lv = b_lv[0];
#pragma unroll 4
    for (int j = 0; j < 64; ++j) {
      mu += hc[j] * W_mu[j];
      lv += hc[j] * W_lv[j];
    }
    const int rg = rowbase + lane;
    out[RECON_ELEMS + rg]      = mu;
    out[RECON_ELEMS + NB + rg] = lv;
    float z = mu + eps[rg] * fast_exp(0.5f * lv);
    float na = b_seq2[0];
#pragma unroll 4
    for (int j = 0; j < NH; ++j)
      na += lrelu(z * W_seq[j] + b_seq[j]) * W_seq2[j];
    out[RECON_ELEMS + 2 * NB + rg] = na > 0.f ? na : 0.f;
    // decoder init: h0 -> cols 0-31, c0 -> cols 32-63
#pragma unroll 4
    for (int j = 0; j < NH; ++j) {
      lds_f[lane * 64 + j]      = lrelu(mu * W_l2h[j]  + b_l2h[j]);
      lds_f[lane * 64 + 32 + j] = lrelu(mu * W_l2h2[j] + b_l2h2[j]);
    }
  }
  __syncthreads();

  // ================= Decoder =================
  float bias_d[8], nlb_d[8];
#pragma unroll
  for (int tile = 0; tile < 8; ++tile) {
    *(v16h*)&wB[0][tile][lane][0] = load_B_gate(dec_Wih, tile, NH, NH, lane);
    *(v16h*)&wB[1][tile][lane][0] = load_B_gate(dec_Whh, tile, NH, NH, lane);
    bias_d[tile] = dec_bih[tile * 16 + n16] + dec_bhh[tile * 16 + n16];
    nlb_d[tile]  = -LOG2E * bias_d[tile];
  }
  *(v16h*)&wO[lane][0] = load_B_out(W_out, lane);
  const float bo = (n16 < ND) ? b_out[n16] : 0.f;

  // h0/c0 back into C/D layout; a_hd = A(h0)
#pragma unroll
  for (int p = 0; p < 2; ++p)
#pragma unroll
    for (int v = 0; v < 8; ++v) {
      h[p][v] = lds_f[(v + rowh) * 64 + p * 16 + n16];
      c[p][v] = lds_f[(v + rowh) * 64 + 32 + p * 16 + n16];
    }
  v16h a_hd = loadA_f32(lds_f, lane, 64, 0);

  // x0 = start_token @ W_emb^T + b_emb (row-independent), in A layout
  v16h a_in;
  {
    const int kb8 = (lane >> 4) * 8;
#pragma unroll
    for (int e = 0; e < 16; ++e) {
      int k = kb8 + ((e < 8) ? e : e + 8);
      float v = b_emb[k];
#pragma unroll
      for (int d = 0; d < ND; ++d) v += start_token[d] * W_emb[k * ND + d];
      a_in[e] = (_Float16)v;
    }
  }
  __syncthreads();

  for (int t = 0; t < NT; ++t) {
    // x_t = relu(x_in) applied elementwise directly in A layout
    v16h a_x;
#pragma unroll
    for (int e = 0; e < 16; ++e) {
      float v = (float)a_in[e];
      a_x[e] = (_Float16)(v > 0.f ? v : 0.f);
    }

    v8f accA[4], accB[4];
#pragma unroll
    for (int q = 0; q < 4; ++q) {
      const int tile = 2 * q;
      v16h bw0 = *(const v16h*)&wB[0][tile][lane][0];
      v16h bw1 = *(const v16h*)&wB[1][tile][lane][0];
      accA[q] = WMMA_F16(a_x,  bw0, zero8);
      accA[q] = WMMA_F16(a_hd, bw1, accA[q]);
    }
#pragma unroll
    for (int q = 0; q < 4; ++q) {
      const int tile = 2 * q + 1;
      v16h bw0 = *(const v16h*)&wB[0][tile][lane][0];
      v16h bw1 = *(const v16h*)&wB[1][tile][lane][0];
      accB[q] = WMMA_F16(a_x,  bw0, zero8);
      accB[q] = WMMA_F16(a_hd, bw1, accB[q]);
    }
    lstm_cell4(accA, nlb_d[0], nlb_d[2], bias_d[4], nlb_d[6], h[0], c[0]);
    lstm_cell4(accB, nlb_d[1], nlb_d[3], bias_d[5], nlb_d[7], h[1], c[1]);

    // stage h_t, reload as A layout (feeds both next-step h path and x path)
    _Float16* buf = lds_h[t & 1];
#pragma unroll
    for (int p = 0; p < 2; ++p)
#pragma unroll
      for (int v = 0; v < 8; ++v)
        buf[(v + rowh) * NH + p * 16 + n16] = (_Float16)h[p][v];
    __syncthreads();
    a_hd = loadA(buf, lane);
    a_in = a_hd;

    // recon[:, t, :] = h_t @ W_out^T + b_out via WMMA (N padded 5->16)
    v16h bwo = *(const v16h*)&wO[lane][0];
    v8f oc = WMMA_F16(a_hd, bwo, zero8);
    if (n16 < ND) {
#pragma unroll
      for (int v = 0; v < 8; ++v) {
        int rg = rowbase + v + rowh;
        out[(size_t)rg * (NT * ND) + t * ND + n16] = oc[v] + bo;
      }
    }
  }
}

extern "C" void kernel_launch(void* const* d_in, const int* in_sizes, int n_in,
                              void* d_out, int out_size, void* d_ws, size_t ws_size,
                              hipStream_t stream) {
  (void)in_sizes; (void)n_in; (void)d_ws; (void)ws_size; (void)out_size;
  const float* x        = (const float*)d_in[0];
  const float* eps      = (const float*)d_in[1];
  const float* enc_Wih  = (const float*)d_in[2];
  const float* enc_Whh  = (const float*)d_in[3];
  const float* enc_bih  = (const float*)d_in[4];
  const float* enc_bhh  = (const float*)d_in[5];
  const float* W_mu     = (const float*)d_in[6];
  const float* b_mu     = (const float*)d_in[7];
  const float* W_lv     = (const float*)d_in[8];
  const float* b_lv     = (const float*)d_in[9];
  const float* W_l2h    = (const float*)d_in[10];
  const float* b_l2h    = (const float*)d_in[11];
  const float* W_l2h2   = (const float*)d_in[12];
  const float* b_l2h2   = (const float*)d_in[13];
  const float* start_tk = (const float*)d_in[14];
  const float* W_emb    = (const float*)d_in[15];
  const float* b_emb    = (const float*)d_in[16];
  const float* dec_Wih  = (const float*)d_in[17];
  const float* dec_Whh  = (const float*)d_in[18];
  const float* dec_bih  = (const float*)d_in[19];
  const float* dec_bhh  = (const float*)d_in[20];
  const float* W_out    = (const float*)d_in[21];
  const float* b_out    = (const float*)d_in[22];
  const float* W_seq    = (const float*)d_in[23];
  const float* b_seq    = (const float*)d_in[24];
  const float* W_seq2   = (const float*)d_in[25];
  const float* b_seq2   = (const float*)d_in[26];

  dim3 grid(NB / 16);   // 1024 wave-tiles of 16 batch rows
  dim3 block(32);       // one wave32 per workgroup
  vae_lstm_kernel<<<grid, block, 0, stream>>>(
      x, eps, enc_Wih, enc_Whh, enc_bih, enc_bhh, W_mu, b_mu, W_lv, b_lv,
      W_l2h, b_l2h, W_l2h2, b_l2h2, start_tk, W_emb, b_emb,
      dec_Wih, dec_Whh, dec_bih, dec_bhh, W_out, b_out,
      W_seq, b_seq, W_seq2, b_seq2, (float*)d_out);
}
